// GAT_62105227100841
// MI455X (gfx1250) — compile-verified, matrix operand order checked by reading, not run on hardware
//
#include <hip/hip_runtime.h>
#include <hip/hip_bf16.h>

#define F_DIM 128
#define NEG_SLOPE 0.2f

typedef __attribute__((ext_vector_type(16))) __bf16 v16bf;
typedef __attribute__((ext_vector_type(8)))  float  v8f;

// packed B fragments per weight matrix: 8 col-tiles * 4 ksteps * 32 lanes * 16 bf16
#define WPACK_ELEMS (8 * 4 * 32 * 16)   // 16384 bf16 = 32 KB

__device__ __forceinline__ float leaky(float z) {
    return z > 0.0f ? z : NEG_SLOPE * z;
}

// float atomic max via signed-int (positive) / unsigned-int (negative) monotonicity
__device__ __forceinline__ void atomicMaxF32(float* addr, float val) {
    if (val >= 0.0f) {
        atomicMax((int*)addr, __float_as_int(val));
    } else {
        atomicMin((unsigned int*)addr, __float_as_uint(val));
    }
}

// Pre-convert W1_src / W2_src (f32, row-major [K=128][N=128]) into bf16 fragments
// laid out exactly as V_WMMA_F32_16X16X32_BF16 expects for B:
//   packed[p][w][t][lane][j],  j<8 : K = 32t + koff + j,  j>=8 : K = 32t+16+koff+(j-8)
//   koff = (lane>>4)*8, col = 16w + (lane&15)
__global__ void gat_pack_w(const float* __restrict__ W1s, const float* __restrict__ W2s,
                           __bf16* __restrict__ packed) {
    int idx  = blockIdx.x * blockDim.x + threadIdx.x;   // 2 * 16384 total
    int j    = idx & 15;
    int lane = (idx >> 4) & 31;
    int t    = (idx >> 9) & 3;
    int w    = (idx >> 11) & 7;
    int p    = (idx >> 14) & 1;
    const float* W = p ? W2s : W1s;
    int koff = (lane >> 4) * 8;
    int mn   = lane & 15;
    int k    = t * 32 + ((j >> 3) << 4) + koff + (j & 7);
    int col  = w * 16 + mn;
    packed[idx] = (__bf16)W[k * F_DIM + col];
}

// v[f] = sum_d W[f,d] * a[d]  for 4 (W,a) pairs (one block each, 128 threads)
__global__ void gat_matvec(const float* __restrict__ W1s, const float* __restrict__ a1s,
                           const float* __restrict__ W1d, const float* __restrict__ a1d,
                           const float* __restrict__ W2s, const float* __restrict__ a2s,
                           const float* __restrict__ W2d, const float* __restrict__ a2d,
                           float* __restrict__ vecs) {
    const float* W; const float* a;
    switch (blockIdx.x) {
        case 0:  W = W1s; a = a1s; break;
        case 1:  W = W1d; a = a1d; break;
        case 2:  W = W2s; a = a2s; break;
        default: W = W2d; a = a2d; break;
    }
    int f = threadIdx.x;
    float s = 0.0f;
    #pragma unroll 8
    for (int d = 0; d < F_DIM; ++d) s += W[f * F_DIM + d] * a[d];
    vecs[blockIdx.x * F_DIM + f] = s;
}

// al_s[n] = x[n,:].v_s ; al_d[n] = x[n,:].v_d  (one wave32 per row, float4 per lane)
__global__ void __launch_bounds__(256)
gat_rowdot(const float* __restrict__ X, const float* __restrict__ vs,
           const float* __restrict__ vd, float* __restrict__ al_s,
           float* __restrict__ al_d, int N) {
    int wave = threadIdx.x >> 5, lane = threadIdx.x & 31;
    int row = blockIdx.x * 8 + wave;
    if (row >= N) return;
    float4 xv = ((const float4*)(X + (size_t)row * F_DIM))[lane];
    float4 av = ((const float4*)vs)[lane];
    float4 bv = ((const float4*)vd)[lane];
    float ss = xv.x * av.x + xv.y * av.y + xv.z * av.z + xv.w * av.w;
    float sd = xv.x * bv.x + xv.y * bv.y + xv.z * bv.z + xv.w * bv.w;
    #pragma unroll
    for (int off = 16; off > 0; off >>= 1) {
        ss += __shfl_down(ss, off, 32);
        sd += __shfl_down(sd, off, 32);
    }
    if (lane == 0) { al_s[row] = ss; al_d[row] = sd; }
}

// C[N,128] = A[N,128] @ B[128,128]  via V_WMMA_F32_16X16X32_BF16.
// Block: 256 thr = 8 waves; block = 16 rows, wave w = cols [16w, 16w+16).
// A tile staged to LDS as bf16; B fragments pre-packed bf16 (32 B contiguous / lane).
__global__ void __launch_bounds__(256)
gat_gemm_wmma(const float* __restrict__ A, const __bf16* __restrict__ Wp,
              float* __restrict__ C, int N) {
    __shared__ __bf16 As[16][F_DIM];   // 4 KB A tile in bf16
    const int tid  = threadIdx.x;
    const int row0 = blockIdx.x * 16;

    // Stage A tile -> LDS (coalesced: 8 consecutive elems per thread, cvt on store)
    #pragma unroll
    for (int j = 0; j < 8; ++j) {
        int i = tid * 8 + j;
        int r = i >> 7, k = i & 127;
        As[r][k] = (__bf16)A[(size_t)(row0 + r) * F_DIM + k];
    }
    __syncthreads();

    const int wave = tid >> 5, lane = tid & 31;
    const int col0 = wave * 16;
    const int mn   = lane & 15;          // A row within tile / C column within tile
    const int koff = (lane >> 4) * 8;    // K sub-offset per half-wave (ISA frag layout)

    const v16bf* Bfrag = (const v16bf*)Wp + (size_t)(wave * 4) * 32 + lane;

    v8f acc = {};
    #pragma unroll
    for (int t = 0; t < 4; ++t) {        // K = 128 in steps of 32
        v16bf af;
        #pragma unroll
        for (int i = 0; i < 8; ++i) {
            af[i]     = As[mn][t * 32 + koff + i];
            af[i + 8] = As[mn][t * 32 + 16 + koff + i];
        }
        v16bf bf = Bfrag[t * 32];        // 32 B contiguous, L2-resident
        acc = __builtin_amdgcn_wmma_f32_16x16x32_bf16(
            /*neg_a=*/false, af, /*neg_b=*/false, bf,
            /*c_mod=*/(short)0, acc, /*reuse_a=*/false, /*reuse_b=*/false);
    }

    const int mb = (lane >> 4) * 8;      // C layout: VGPR v -> row mb+v, col mn
    #pragma unroll
    for (int v = 0; v < 8; ++v)
        C[(size_t)(row0 + mb + v) * F_DIM + col0 + mn] = acc[v];
}

// zero accumulator, init segment-max to -FLT_MAX, denom to 0
__global__ void gat_init(float* __restrict__ m, float* __restrict__ denom,
                         float* __restrict__ acc, int N) {
    int i = blockIdx.x * blockDim.x + threadIdx.x;
    if (i < N * F_DIM) acc[i] = 0.0f;
    if (i < N) { m[i] = -3.402823466e38f; denom[i] = 0.0f; }
}

__global__ void gat_edge_max(const int* __restrict__ src, const int* __restrict__ dst,
                             const float* __restrict__ al_s, const float* __restrict__ al_d,
                             float* __restrict__ m, int E) {
    int e = blockIdx.x * blockDim.x + threadIdx.x;
    if (e >= E) return;
    int d = dst[e];
    float z = leaky(al_s[src[e]] + al_d[d]);
    atomicMaxF32(m + d, z);
}

__global__ void gat_edge_exp(const int* __restrict__ src, const int* __restrict__ dst,
                             const float* __restrict__ al_s, const float* __restrict__ al_d,
                             const float* __restrict__ m, float* __restrict__ ebuf,
                             float* __restrict__ denom, int E) {
    int e = blockIdx.x * blockDim.x + threadIdx.x;
    if (e >= E) return;
    int d = dst[e];
    float z  = leaky(al_s[src[e]] + al_d[d]);
    float ex = __expf(z - m[d]);
    ebuf[e] = ex;
    atomicAdd(denom + d, ex);
}

// acc[dst,:] += (e/denom[dst]) * hs[src,:]  — one wave32 per edge, float4 per lane.
// hs + acc are L2-resident (51 MB each vs 192 MB L2) -> atomics run at L2 rate.
__global__ void __launch_bounds__(256)
gat_edge_agg(const int* __restrict__ src, const int* __restrict__ dst,
             const float* __restrict__ hs, const float* __restrict__ ebuf,
             const float* __restrict__ denom, float* __restrict__ acc, int E) {
    int wave = threadIdx.x >> 5, lane = threadIdx.x & 31;
    int e = blockIdx.x * 8 + wave;
    if (e >= E) return;
    int s = src[e], d = dst[e];
    float alpha = ebuf[e] / denom[d];
    float4 hv = ((const float4*)(hs + (size_t)s * F_DIM))[lane];
    float* out = acc + (size_t)d * F_DIM + lane * 4;
    atomicAdd(out + 0, alpha * hv.x);
    atomicAdd(out + 1, alpha * hv.y);
    atomicAdd(out + 2, alpha * hv.z);
    atomicAdd(out + 3, alpha * hv.w);
}

__global__ void gat_finalize(const float* __restrict__ acc, const float* __restrict__ bias,
                             float* __restrict__ out, int N, int do_relu) {
    int i = blockIdx.x * blockDim.x + threadIdx.x;
    if (i >= N * F_DIM) return;
    float v = acc[i] + bias[i & (F_DIM - 1)];
    if (do_relu) v = fmaxf(v, 0.0f);
    out[i] = v;
}

extern "C" void kernel_launch(void* const* d_in, const int* in_sizes, int n_in,
                              void* d_out, int out_size, void* d_ws, size_t ws_size,
                              hipStream_t stream) {
    const float* x   = (const float*)d_in[0];
    const int*   ei  = (const int*)d_in[1];
    const float* W1s = (const float*)d_in[2];
    const float* W1d = (const float*)d_in[3];
    const float* a1s = (const float*)d_in[4];
    const float* a1d = (const float*)d_in[5];
    const float* b1  = (const float*)d_in[6];
    const float* W2s = (const float*)d_in[7];
    const float* W2d = (const float*)d_in[8];
    const float* a2s = (const float*)d_in[9];
    const float* a2d = (const float*)d_in[10];
    const float* b2  = (const float*)d_in[11];

    const int N = in_sizes[0] / F_DIM;
    const int E = in_sizes[1] / 2;
    const int* src = ei;
    const int* dst = ei + E;

    // ws layout: packed bf16 W fragments first (32 B-aligned vector loads), then f32.
    __bf16* wpack = (__bf16*)d_ws;                       // 2 * WPACK_ELEMS bf16 = 64 KB
    float*  fws   = (float*)((char*)d_ws + 2 * WPACK_ELEMS * sizeof(__bf16));
    float* hs    = fws;
    float* hbuf  = hs   + (size_t)N * F_DIM;
    float* al_s  = hbuf + (size_t)N * F_DIM;
    float* al_d  = al_s + N;
    float* mbuf  = al_d + N;
    float* denom = mbuf + N;
    float* ebuf  = denom + N;
    float* vecs  = ebuf + E;   // 4 * 128 precomputed W@a vectors

    float* out = (float*)d_out;

    dim3 b256(256);
    const int gNF   = (N * F_DIM + 255) / 256;
    const int gE    = (E + 255) / 256;
    const int gAgg  = (E + 7) / 8;
    const int gRow  = (N + 7) / 8;
    const int gGemm = (N + 15) / 16;   // N = 100000 -> exact

    // one-shot prep: pack W_src matrices into WMMA B-fragment layout; W@a matvecs
    gat_pack_w<<<(2 * WPACK_ELEMS + 255) / 256, b256, 0, stream>>>(W1s, W2s, wpack);
    gat_matvec<<<4, 128, 0, stream>>>(W1s, a1s, W1d, a1d, W2s, a2s, W2d, a2d, vecs);

    // ---- layer 1 ----
    gat_rowdot   <<<gRow,  b256, 0, stream>>>(x, vecs, vecs + 128, al_s, al_d, N);
    gat_gemm_wmma<<<gGemm, b256, 0, stream>>>(x, wpack, hs, N);
    gat_init     <<<gNF,   b256, 0, stream>>>(mbuf, denom, hbuf, N);
    gat_edge_max <<<gE,    b256, 0, stream>>>(src, dst, al_s, al_d, mbuf, E);
    gat_edge_exp <<<gE,    b256, 0, stream>>>(src, dst, al_s, al_d, mbuf, ebuf, denom, E);
    gat_edge_agg <<<gAgg,  b256, 0, stream>>>(src, dst, hs, ebuf, denom, hbuf, E);
    gat_finalize <<<gNF,   b256, 0, stream>>>(hbuf, b1, hbuf, N, 1);

    // ---- layer 2 ----
    gat_rowdot   <<<gRow,  b256, 0, stream>>>(hbuf, vecs + 256, vecs + 384, al_s, al_d, N);
    gat_gemm_wmma<<<gGemm, b256, 0, stream>>>(hbuf, wpack + WPACK_ELEMS, hs, N);
    gat_init     <<<gNF,   b256, 0, stream>>>(mbuf, denom, out, N);
    gat_edge_max <<<gE,    b256, 0, stream>>>(src, dst, al_s, al_d, mbuf, E);
    gat_edge_exp <<<gE,    b256, 0, stream>>>(src, dst, al_s, al_d, mbuf, ebuf, denom, E);
    gat_edge_agg <<<gAgg,  b256, 0, stream>>>(src, dst, hs, ebuf, denom, out, E);
    gat_finalize <<<gNF,   b256, 0, stream>>>(out, b2, out, N, 0);
}
